// SH_Attention_Module_68453188764060
// MI455X (gfx1250) — matrix operand order repaired and checked by reading, not compiled, not run on hardware
//
#include <hip/hip_runtime.h>
#include <hip/hip_bf16.h>
#include <math.h>

// ---------------------------------------------------------------------------
// Types for CDNA5 WMMA (wave32): A/B = 16 bf16 (8 VGPRs), C/D = 8 f32 (8 VGPRs)
// ---------------------------------------------------------------------------
typedef __attribute__((ext_vector_type(16))) __bf16 bf16x16;
typedef __attribute__((ext_vector_type(8)))  float  f32x8;

union AFrag { bf16x16 v; unsigned short u[16]; uint4 q[2]; };

#define NSH 55

__device__ __forceinline__ unsigned short f2bf(float f) {
  unsigned int u = __float_as_uint(f);
  unsigned int r = u + 0x7FFFu + ((u >> 16) & 1u);   // round-to-nearest-even
  return (unsigned short)(r >> 16);
}

__device__ __forceinline__ float geluf(float x) {
  return 0.5f * x * (1.0f + erff(x * 0.70710678118654752440f));
}

// ---------------------------------------------------------------------------
// Elementwise cast f32 -> bf16 (bit pattern in ushort buffers)
// ---------------------------------------------------------------------------
__global__ void k_cast_bf16(const float* __restrict__ x, unsigned short* __restrict__ y, size_t n) {
  size_t i = (size_t)blockIdx.x * blockDim.x + threadIdx.x;
  size_t stride = (size_t)gridDim.x * blockDim.x;
  for (; i < n; i += stride) y[i] = f2bf(x[i]);
}

// Rearrange conv weight [128,256,3,3] f32 -> [9 taps][128 oc][256 ic] bf16
__global__ void k_rearrange_w3x3(const float* __restrict__ w, unsigned short* __restrict__ o) {
  int i = blockIdx.x * blockDim.x + threadIdx.x;
  if (i >= 128 * 256 * 9) return;
  int tap = i % 9;
  int ic  = (i / 9) % 256;
  int oc  = i / (9 * 256);
  o[((size_t)tap * 128 + oc) * 256 + ic] = f2bf(w[((size_t)oc * 256 + ic) * 9 + tap]);
}

// im2col for non-overlapping patch conv (stride == kernel), bf16 output [K, P]
__global__ void k_im2col_patch(const float* __restrict__ x, unsigned short* __restrict__ col,
                               int IC, int H, int W, int S) {
  int PW = W / S, PH = H / S;
  int P = PH * PW;
  size_t K = (size_t)IC * S * S;
  size_t total = K * (size_t)P;
  size_t i = (size_t)blockIdx.x * blockDim.x + threadIdx.x;
  size_t stride = (size_t)gridDim.x * blockDim.x;
  for (; i < total; i += stride) {
    int p = (int)(i % P);
    size_t k = i / P;
    int kw = (int)(k % S);
    int kh = (int)((k / S) % S);
    int ic = (int)(k / ((size_t)S * S));
    int py = p / PW, px = p % PW;
    col[i] = f2bf(x[((size_t)ic * H + (size_t)py * S + kh) * W + (size_t)px * S + kw]);
  }
}

// ---------------------------------------------------------------------------
// Fragment-major LDS slot for the B tile:
//   element (k, n) of the 32x64 tile lives at BsF[((n>>4)<<5)|(k&16)|(n&15)][k&15]
//   so each lane's 16 bf16 B-fragment values are 32 contiguous bytes.
// ---------------------------------------------------------------------------
__device__ __forceinline__ int bslot(int k, int n) {
  return ((n >> 4) << 5) | (k & 16) | (n & 15);
}

// ---------------------------------------------------------------------------
// Generic bf16 WMMA GEMM: C[M,N] = act(alpha*A@B(+bias)) (+residual)
//   A: [M,K] row-major bf16, B: [K,N] row-major (or [N,K] if transB)
//   128 threads = 4 waves; tile 64(M) x 64(N) x 32(K); LDS staged.
// ---------------------------------------------------------------------------
__global__ void __launch_bounds__(128)
k_gemm_bf16(const unsigned short* __restrict__ A,
            const unsigned short* __restrict__ B,
            const float* __restrict__ bias, int biasMode,           // 0 none, 1 per-col, 2 per-row
            const float* __restrict__ residual,
            float* __restrict__ C,
            int M, int N, int K, int transB, float alpha, int act)  // act: 0 none, 1 gelu
{
  __shared__ __align__(16) unsigned short As[64][32];    // [m][k] plain
  __shared__ __align__(16) unsigned short BsF[128][16];  // fragment-major
  const int tid  = threadIdx.x;
  const int wave = tid >> 5;
  const int lane = tid & 31;
  const int half = lane >> 4;
  const int l16  = lane & 15;
  const int bm = blockIdx.x * 64;
  const int bn = blockIdx.y * 64;

  f32x8 acc[4];
  #pragma unroll
  for (int t = 0; t < 4; ++t)
    #pragma unroll
    for (int j = 0; j < 8; ++j) acc[t][j] = 0.0f;

  for (int k0 = 0; k0 < K; k0 += 32) {
    if (k0 + 32 < K) __builtin_prefetch(A + (size_t)bm * K + k0 + 32, 0, 1);
    for (int i = tid; i < 64 * 32; i += 128) {
      int r = i >> 5, c = i & 31;
      int gm = bm + r, gk = k0 + c;
      As[r][c] = (gm < M && gk < K) ? A[(size_t)gm * K + gk] : (unsigned short)0;
    }
    for (int i = tid; i < 32 * 64; i += 128) {
      int r = i >> 6, c = i & 63;   // r = k, c = n
      int gk = k0 + r, gn = bn + c;
      unsigned short v = 0;
      if (gk < K && gn < N)
        v = transB ? B[(size_t)gn * K + gk] : B[(size_t)gk * N + gn];
      BsF[bslot(r, c)][r & 15] = v;
    }
    __syncthreads();

    // A fragment: two contiguous 16B runs of row (wave*16+l16)
    AFrag fa;
    {
      const unsigned short* ap = &As[wave * 16 + l16][0];
      fa.q[0] = *(const uint4*)(ap + half * 16);
      fa.q[1] = *(const uint4*)(ap + half * 16 + 8);
    }
    #pragma unroll
    for (int t = 0; t < 4; ++t) {
      AFrag fb;   // one contiguous 32B chunk per lane
      const unsigned short* bp = &BsF[(t << 5) | lane][0];
      fb.q[0] = *(const uint4*)bp;
      fb.q[1] = *(const uint4*)(bp + 8);
      acc[t] = __builtin_amdgcn_wmma_f32_16x16x32_bf16(false, fa.v, false, fb.v,
                                                       (short)0, acc[t], false, false);
    }
    __syncthreads();
  }

  #pragma unroll
  for (int t = 0; t < 4; ++t) {
    #pragma unroll
    for (int j = 0; j < 8; ++j) {
      int gm = bm + wave * 16 + j + 8 * half;   // C layout: VGPR j, lane-half selects M +8
      int gn = bn + t * 16 + l16;
      if (gm < M && gn < N) {
        float x = acc[t][j] * alpha;
        if (biasMode == 1)      x += bias[gn];
        else if (biasMode == 2) x += bias[gm];
        if (act == 1) x = geluf(x);
        if (residual) x += residual[(size_t)gm * N + gn];
        C[(size_t)gm * N + gn] = x;
      }
    }
  }
}

// ---------------------------------------------------------------------------
// 3x3 SAME conv as implicit GEMM (9 shifted taps), bf16 WMMA, f32 out.
//   x: [B,256,H,W] bf16, w: [9,128,256] bf16, out: [B,128,H,W] f32
//   256 threads = 8 waves; tile 128(OC) x 64(pixels of one row) x 32(IC)
// ---------------------------------------------------------------------------
__global__ void __launch_bounds__(256)
k_conv3x3_wmma(const unsigned short* __restrict__ xbf,
               const unsigned short* __restrict__ wbf,
               float* __restrict__ out, int H, int W)
{
  __shared__ __align__(16) unsigned short As[128][32];
  __shared__ __align__(16) unsigned short BsF[128][16];
  const int tid  = threadIdx.x;
  const int wave = tid >> 5;
  const int lane = tid & 31;
  const int half = lane >> 4;
  const int l16  = lane & 15;
  const int tilesPerRow = W >> 6;
  const int b  = blockIdx.y;
  const int y  = blockIdx.x / tilesPerRow;
  const int x0 = (blockIdx.x % tilesPerRow) * 64;
  const size_t plane = (size_t)H * W;
  const unsigned short* xb = xbf + (size_t)b * 256 * plane;

  f32x8 acc[4];
  #pragma unroll
  for (int t = 0; t < 4; ++t)
    #pragma unroll
    for (int j = 0; j < 8; ++j) acc[t][j] = 0.0f;

  for (int kh = 0; kh < 3; ++kh) {
    const int yy = y + kh - 1;
    const bool rowOK = (yy >= 0) && (yy < H);
    for (int kw = 0; kw < 3; ++kw) {
      const unsigned short* wt = wbf + (size_t)(kh * 3 + kw) * 128 * 256;
      for (int k0 = 0; k0 < 256; k0 += 32) {
        // A tile (weights): rows are 16B-aligned bf16 -> uint4 copies
        for (int i = tid; i < 512; i += 256) {        // 512 chunks of 8 ushorts
          int r = i >> 2, c8 = (i & 3) * 8;
          *(uint4*)&As[r][c8] = *(const uint4*)&wt[(size_t)r * 256 + k0 + c8];
        }
        // B tile (shifted input row), staged fragment-major with bounds masking
        for (int i = tid; i < 32 * 64; i += 256) {
          int r = i >> 6, c = i & 63;
          int xx = x0 + c + kw - 1;
          unsigned short v = 0;
          if (rowOK && xx >= 0 && xx < W)
            v = xb[(size_t)(k0 + r) * plane + (size_t)yy * W + xx];
          BsF[bslot(r, c)][r & 15] = v;
        }
        __syncthreads();
        AFrag fa;
        {
          const unsigned short* ap = &As[wave * 16 + l16][0];
          fa.q[0] = *(const uint4*)(ap + half * 16);
          fa.q[1] = *(const uint4*)(ap + half * 16 + 8);
        }
        #pragma unroll
        for (int t = 0; t < 4; ++t) {
          AFrag fb;
          const unsigned short* bp = &BsF[(t << 5) | lane][0];
          fb.q[0] = *(const uint4*)bp;
          fb.q[1] = *(const uint4*)(bp + 8);
          acc[t] = __builtin_amdgcn_wmma_f32_16x16x32_bf16(false, fa.v, false, fb.v,
                                                           (short)0, acc[t], false, false);
        }
        __syncthreads();
      }
    }
  }

  #pragma unroll
  for (int t = 0; t < 4; ++t) {
    #pragma unroll
    for (int j = 0; j < 8; ++j) {
      int oc = wave * 16 + j + 8 * half;
      int px = x0 + t * 16 + l16;
      out[((size_t)b * 128 + oc) * plane + (size_t)y * W + px] = acc[t][j];
    }
  }
}

// ---------------------------------------------------------------------------
// LayerNorm over last dim (=128), one block (128 threads) per row
// ---------------------------------------------------------------------------
__global__ void __launch_bounds__(128)
k_layernorm(const float* __restrict__ x, const float* __restrict__ g,
            const float* __restrict__ b, float* __restrict__ y, int rows)
{
  __shared__ float red[128];
  int r = blockIdx.x;
  int t = threadIdx.x;
  float v = x[(size_t)r * 128 + t];
  red[t] = v; __syncthreads();
  for (int s = 64; s > 0; s >>= 1) { if (t < s) red[t] += red[t + s]; __syncthreads(); }
  float m = red[0] * (1.0f / 128.0f);
  __syncthreads();
  red[t] = v * v; __syncthreads();
  for (int s = 64; s > 0; s >>= 1) { if (t < s) red[t] += red[t + s]; __syncthreads(); }
  float var = red[0] * (1.0f / 128.0f) - m * m;
  y[(size_t)r * 128 + t] = (v - m) * rsqrtf(var + 1e-5f) * g[t] + b[t];
  (void)rows;
}

// Softmax over the row(query) axis: per column (one block each), rows entries
__global__ void __launch_bounds__(256)
k_col_softmax(float* __restrict__ S, int rows, int cols)
{
  __shared__ float red[256];
  int c = blockIdx.x, t = threadIdx.x;
  float mx = -3.4e38f;
  for (int r = t; r < rows; r += 256) mx = fmaxf(mx, S[(size_t)r * cols + c]);
  red[t] = mx; __syncthreads();
  for (int s = 128; s > 0; s >>= 1) { if (t < s) red[t] = fmaxf(red[t], red[t + s]); __syncthreads(); }
  mx = red[0]; __syncthreads();
  float sum = 0.0f;
  for (int r = t; r < rows; r += 256) sum += expf(S[(size_t)r * cols + c] - mx);
  red[t] = sum; __syncthreads();
  for (int s = 128; s > 0; s >>= 1) { if (t < s) red[t] += red[t + s]; __syncthreads(); }
  float inv = 1.0f / red[0];
  for (int r = t; r < rows; r += 256) {
    size_t i = (size_t)r * cols + c;
    S[i] = expf(S[i] - mx) * inv;
  }
}

// Renormalize each row: a /= (sum_k a + 1e-8)
__global__ void k_row_renorm(float* __restrict__ S, int rows, int cols) {
  int r = blockIdx.x * blockDim.x + threadIdx.x;
  if (r >= rows) return;
  float s = 0.0f;
  for (int c = 0; c < cols; ++c) s += S[(size_t)r * cols + c];
  float inv = 1.0f / (s + 1e-8f);
  for (int c = 0; c < cols; ++c) S[(size_t)r * cols + c] *= inv;
}

// L2 norm of each 128-dim row
__global__ void __launch_bounds__(128)
k_row_l2norm(const float* __restrict__ x, float* __restrict__ n, int rows)
{
  __shared__ float red[128];
  int r = blockIdx.x, t = threadIdx.x;
  float v = x[(size_t)r * 128 + t];
  red[t] = v * v; __syncthreads();
  for (int s = 64; s > 0; s >>= 1) { if (t < s) red[t] += red[t + s]; __syncthreads(); }
  if (t == 0) n[r] = sqrtf(red[0]);
  (void)rows;
}

// Cosine-sim argmax over 55 SH tokens per pixel, then pf *= SH[argmax]
// shq: [55,128] (per batch), shn: [55], pf: [128,HW] modified in place
__global__ void __launch_bounds__(256)
k_simgather(const float* __restrict__ shq, const float* __restrict__ shn,
            float* __restrict__ pf, int HWn)
{
  __shared__ float sh[NSH * 128];
  __shared__ float sn[NSH];
  int t = threadIdx.x;
  for (int i = t; i < NSH * 128; i += 256) sh[i] = shq[i];
  if (t < NSH) sn[t] = shn[t];
  __syncthreads();
  for (int p = blockIdx.x * 256 + t; p < HWn; p += gridDim.x * 256) {
    float acc[NSH];
    #pragma unroll
    for (int n2 = 0; n2 < NSH; ++n2) acc[n2] = 0.0f;
    for (int c = 0; c < 128; ++c) {
      float v = pf[(size_t)c * HWn + p];
      #pragma unroll
      for (int n2 = 0; n2 < NSH; ++n2) acc[n2] += v * sh[n2 * 128 + c];
    }
    int best = 0;
    float bv = acc[0] / sn[0];
    #pragma unroll
    for (int n2 = 1; n2 < NSH; ++n2) {
      float f = acc[n2] / sn[n2];
      if (f > bv) { bv = f; best = n2; }   // strict >: first-max like jnp.argmax
    }
    for (int c = 0; c < 128; ++c)
      pf[(size_t)c * HWn + p] *= sh[best * 128 + c];
  }
}

// Tiled transpose: in [R,C] -> out [C,R]
__global__ void k_transpose(const float* __restrict__ in, float* __restrict__ out, int R, int Cc) {
  __shared__ float tile[32][33];
  int x = blockIdx.x * 32 + threadIdx.x;
  int y0 = blockIdx.y * 32;
  for (int j = threadIdx.y; j < 32; j += 8) {
    int y = y0 + j;
    if (y < R && x < Cc) tile[j][threadIdx.x] = in[(size_t)y * Cc + x];
  }
  __syncthreads();
  int ox = y0 + threadIdx.x;
  int oy0 = blockIdx.x * 32;
  for (int j = threadIdx.y; j < 32; j += 8) {
    int oy = oy0 + j;
    if (oy < Cc && ox < R) out[(size_t)oy * R + ox] = tile[threadIdx.x][j];
  }
}

// normal head: t [HW,3] -> out [3,HW] normalized per pixel
__global__ void k_finalize_normal(const float* __restrict__ t, float* __restrict__ o, int HWn) {
  int p = blockIdx.x * blockDim.x + threadIdx.x;
  if (p >= HWn) return;
  float a = t[(size_t)p * 3], b = t[(size_t)p * 3 + 1], c = t[(size_t)p * 3 + 2];
  float inv = rsqrtf(a * a + b * b + c * c);
  o[p] = a * inv; o[HWn + p] = b * inv; o[(size_t)2 * HWn + p] = c * inv;
}

// ---------------------------------------------------------------------------
// Host orchestration
// ---------------------------------------------------------------------------
extern "C" void kernel_launch(void* const* d_in, const int* in_sizes, int n_in,
                              void* d_out, int out_size, void* d_ws, size_t ws_size,
                              hipStream_t stream) {
  (void)in_sizes; (void)n_in; (void)out_size; (void)ws_size;

  // Inputs in setup_inputs() insertion order (nested dicts depth-first)
  const float* in_SH    = (const float*)d_in[0];   // [2,55,256,512]
  const float* in_f0    = (const float*)d_in[1];   // [2,256,32,64]
  const float* in_f1    = (const float*)d_in[2];   // [2,256,64,128]
  const float* in_f2    = (const float*)d_in[3];   // [2,256,128,256]
  const float* p_sh_w   = (const float*)d_in[4];
  const float* p_sh_b   = (const float*)d_in[5];
  const float* p_f0_w   = (const float*)d_in[6];
  const float* p_f0_b   = (const float*)d_in[7];
  const float* p_f1_w   = (const float*)d_in[8];
  const float* p_f1_b   = (const float*)d_in[9];
  const float* p_f2_w   = (const float*)d_in[10];
  const float* p_f2_b   = (const float*)d_in[11];
  const float* p_cd_w   = (const float*)d_in[12];
  const float* p_cn_w   = (const float*)d_in[13];
  const float* p_pe     = (const float*)d_in[14];
  const float* ca_lnsg  = (const float*)d_in[15];
  const float* ca_lnsb  = (const float*)d_in[16];
  const float* ca_lntg  = (const float*)d_in[17];
  const float* ca_lntb  = (const float*)d_in[18];
  const float* ca_wq    = (const float*)d_in[19];
  const float* ca_bq    = (const float*)d_in[20];
  const float* ca_wk    = (const float*)d_in[21];
  const float* ca_bk    = (const float*)d_in[22];
  const float* ca_wv    = (const float*)d_in[23];
  const float* ca_bv    = (const float*)d_in[24];
  const float* ca_wo    = (const float*)d_in[25];
  const float* ca_bo    = (const float*)d_in[26];
  const float* md_lng   = (const float*)d_in[27];
  const float* md_lnb   = (const float*)d_in[28];
  const float* md_w1    = (const float*)d_in[29];
  const float* md_b1    = (const float*)d_in[30];
  const float* md_w2    = (const float*)d_in[31];
  const float* md_b2    = (const float*)d_in[32];
  const float* mn_lng   = (const float*)d_in[33];
  const float* mn_lnb   = (const float*)d_in[34];
  const float* mn_w1    = (const float*)d_in[35];
  const float* mn_b1    = (const float*)d_in[36];
  const float* mn_w2    = (const float*)d_in[37];
  const float* mn_b2    = (const float*)d_in[38];
  const float* pd_ln1g  = (const float*)d_in[39];
  const float* pd_ln1b  = (const float*)d_in[40];
  const float* pd_w1    = (const float*)d_in[41];
  const float* pd_b1    = (const float*)d_in[42];
  const float* pd_ln2g  = (const float*)d_in[43];
  const float* pd_ln2b  = (const float*)d_in[44];
  const float* pd_w2    = (const float*)d_in[45];
  const float* pd_b2    = (const float*)d_in[46];
  const float* pn_ln1g  = (const float*)d_in[47];
  const float* pn_ln1b  = (const float*)d_in[48];
  const float* pn_w1    = (const float*)d_in[49];
  const float* pn_b1    = (const float*)d_in[50];
  const float* pn_ln2g  = (const float*)d_in[51];
  const float* pn_ln2b  = (const float*)d_in[52];
  const float* pn_w2    = (const float*)d_in[53];
  const float* pn_b2    = (const float*)d_in[54];

  float* outF = (float*)d_out;
  const int HW = 32768;   // 128*256

  // ---- workspace bump allocator ----
  size_t off = 0;
  char* base = (char*)d_ws;
  auto allocF = [&](size_t n) -> float* {
    float* p = (float*)(base + off); off += ((n * 4 + 255) & ~(size_t)255); return p;
  };
  auto allocU = [&](size_t n) -> unsigned short* {
    unsigned short* p = (unsigned short*)(base + off); off += ((n * 2 + 255) & ~(size_t)255); return p;
  };

  float* shq   = allocF(2 * 55 * 128);
  float* f0a   = allocF(2 * 128 * 128);
  float* f1a   = allocF(2 * 128 * 128);
  float* f2a   = allocF(2 * 128 * 128);
  float* sh1   = allocF(2 * 55 * 128);
  float* sh2   = allocF(2 * 55 * 128);
  float* sh3   = allocF(2 * 55 * 128);
  float* F1b   = allocF(2 * 128 * 128);
  float* F2b   = allocF(2 * 128 * 128);
  float* shqn  = allocF(2 * 55);

  unsigned short* wq_bf  = allocU(128 * 128);
  unsigned short* wk_bf  = allocU(128 * 128);
  unsigned short* wv_bf  = allocU(128 * 128);
  unsigned short* wo_bf  = allocU(128 * 128);
  unsigned short* mdw1_bf = allocU(128 * 512);
  unsigned short* mdw2_bf = allocU(512 * 128);
  unsigned short* mnw1_bf = allocU(128 * 512);
  unsigned short* mnw2_bf = allocU(512 * 128);
  unsigned short* pdw1_bf = allocU(128 * 128);
  unsigned short* pdw2_bf = allocU(128 * 1);
  unsigned short* pnw1_bf = allocU(128 * 128);
  unsigned short* pnw2_bf = allocU(128 * 3);
  unsigned short* shw_bf  = allocU((size_t)55 * 56320);
  unsigned short* f0w_bf  = allocU((size_t)128 * 4096);
  unsigned short* f1w_bf  = allocU((size_t)128 * 16384);
  unsigned short* f2w_bf  = allocU((size_t)128 * 65536);
  unsigned short* cdw_bf  = allocU(9 * 128 * 256);
  unsigned short* cnw_bf  = allocU(9 * 128 * 256);
  unsigned short* feat2_bf = allocU((size_t)2 * 256 * HW);
  unsigned short* colbuf   = allocU((size_t)65536 * 128);

  float* pfd = allocF((size_t)2 * 128 * HW);
  float* pfn = allocF((size_t)2 * 128 * HW);

  // attention / head scratch (sized for Ns = 32768)
  float* sln = allocF((size_t)HW * 128);         unsigned short* sln_bf = allocU((size_t)HW * 128);
  float* tln = allocF(128 * 128);                unsigned short* tln_bf = allocU(128 * 128);
  float* qb  = allocF((size_t)HW * 128);         unsigned short* qb_bf  = allocU((size_t)HW * 128);
  float* kb  = allocF(128 * 128);                unsigned short* kb_bf  = allocU(128 * 128);
  float* vb  = allocF(128 * 128);                unsigned short* vb_bf  = allocU(128 * 128);
  float* Sb  = allocF((size_t)HW * 128);         unsigned short* Sb_bf  = allocU((size_t)HW * 128);
  float* ob  = allocF((size_t)HW * 128);         unsigned short* ob_bf  = allocU((size_t)HW * 128);
  float* fbuf = allocF((size_t)HW * 128);
  float* hbuf = allocF((size_t)HW * 512);        unsigned short* hbuf_bf = allocU((size_t)HW * 512);
  float* pbuf = allocF((size_t)HW * 128);
  float* obuf = allocF((size_t)HW * 4);

  // ---- helpers ----
  auto cast = [&](const float* x, unsigned short* y, size_t n) {
    int blocks = (int)((n + 255) / 256); if (blocks > 8192) blocks = 8192;
    k_cast_bf16<<<blocks, 256, 0, stream>>>(x, y, n);
  };
  auto gemm = [&](const unsigned short* A, const unsigned short* B, const float* bias, int biasMode,
                  const float* residual, float* C, int M, int N, int K, int transB,
                  float alpha, int act) {
    dim3 grid((M + 63) / 64, (N + 63) / 64);
    k_gemm_bf16<<<grid, 128, 0, stream>>>(A, B, bias, biasMode, residual, C,
                                          M, N, K, transB, alpha, act);
  };
  auto lnorm = [&](const float* x, const float* g, const float* b, float* y, int rows) {
    k_layernorm<<<rows, 128, 0, stream>>>(x, g, b, y, rows);
  };

  // ---- weight casts ----
  cast(ca_wq, wq_bf, 16384);  cast(ca_wk, wk_bf, 16384);
  cast(ca_wv, wv_bf, 16384);  cast(ca_wo, wo_bf, 16384);
  cast(md_w1, mdw1_bf, 65536); cast(md_w2, mdw2_bf, 65536);
  cast(mn_w1, mnw1_bf, 65536); cast(mn_w2, mnw2_bf, 65536);
  cast(pd_w1, pdw1_bf, 16384); cast(pd_w2, pdw2_bf, 128);
  cast(pn_w1, pnw1_bf, 16384); cast(pn_w2, pnw2_bf, 384);
  cast(p_sh_w, shw_bf, (size_t)55 * 56320);
  cast(p_f0_w, f0w_bf, (size_t)128 * 4096);
  cast(p_f1_w, f1w_bf, (size_t)128 * 16384);
  cast(p_f2_w, f2w_bf, (size_t)128 * 65536);
  k_rearrange_w3x3<<<(294912 + 255) / 256, 256, 0, stream>>>(p_cd_w, cdw_bf);
  k_rearrange_w3x3<<<(294912 + 255) / 256, 256, 0, stream>>>(p_cn_w, cnw_bf);
  cast(in_f2, feat2_bf, (size_t)2 * 256 * HW);

  // ---- patch-embed convs (im2col + WMMA GEMM) ----
  for (int b = 0; b < 2; ++b) {
    // sh: [55, 55*32*32] x [K,128] + per-row bias
    k_im2col_patch<<<8192, 256, 0, stream>>>(in_SH + (size_t)b * 55 * 256 * 512, colbuf, 55, 256, 512, 32);
    gemm(shw_bf, colbuf, p_sh_b, 2, nullptr, shq + (size_t)b * 7040, 55, 128, 56320, 0, 1.0f, 0);
    // f0: stride 4, + pe
    k_im2col_patch<<<2048, 256, 0, stream>>>(in_f0 + (size_t)b * 524288, colbuf, 256, 32, 64, 4);
    gemm(f0w_bf, colbuf, p_f0_b, 2, p_pe, f0a + (size_t)b * 16384, 128, 128, 4096, 0, 1.0f, 0);
    // f1: stride 8, + pe
    k_im2col_patch<<<8192, 256, 0, stream>>>(in_f1 + (size_t)b * 2097152, colbuf, 256, 64, 128, 8);
    gemm(f1w_bf, colbuf, p_f1_b, 2, p_pe, f1a + (size_t)b * 16384, 128, 128, 16384, 0, 1.0f, 0);
    // f2: stride 16, + pe
    k_im2col_patch<<<8192, 256, 0, stream>>>(in_f2 + (size_t)b * 8388608, colbuf, 256, 128, 256, 16);
    gemm(f2w_bf, colbuf, p_f2_b, 2, p_pe, f2a + (size_t)b * 16384, 128, 128, 65536, 0, 1.0f, 0);
  }

  // ---- shared cross-attention (sink-competition softmax) ----
  auto attention = [&](const float* sink, int Ns, const float* src, int Nt,
                       const float* resid, float* out) {
    lnorm(sink, ca_lnsg, ca_lnsb, sln, Ns);
    cast(sln, sln_bf, (size_t)Ns * 128);
    lnorm(src, ca_lntg, ca_lntb, tln, Nt);
    cast(tln, tln_bf, (size_t)Nt * 128);
    gemm(sln_bf, wq_bf, ca_bq, 1, nullptr, qb, Ns, 128, 128, 0, 1.0f, 0);
    cast(qb, qb_bf, (size_t)Ns * 128);
    gemm(tln_bf, wk_bf, ca_bk, 1, nullptr, kb, Nt, 128, 128, 0, 1.0f, 0);
    cast(kb, kb_bf, (size_t)Nt * 128);
    gemm(tln_bf, wv_bf, ca_bv, 1, nullptr, vb, Nt, 128, 128, 0, 1.0f, 0);
    cast(vb, vb_bf, (size_t)Nt * 128);
    // scores = q @ k^T / sqrt(128)
    gemm(qb_bf, kb_bf, nullptr, 0, nullptr, Sb, Ns, Nt, 128, 1, 0.08838834764831845f, 0);
    k_col_softmax<<<Nt, 256, 0, stream>>>(Sb, Ns, Nt);        // softmax over query axis
    k_row_renorm<<<(Ns + 255) / 256, 256, 0, stream>>>(Sb, Ns, Nt);
    cast(Sb, Sb_bf, (size_t)Ns * Nt);
    gemm(Sb_bf, vb_bf, nullptr, 0, nullptr, ob, Ns, 128, Nt, 0, 1.0f, 0);
    cast(ob, ob_bf, (size_t)Ns * 128);
    gemm(ob_bf, wo_bf, ca_bo, 1, resid, out, Ns, 128, 128, 0, 1.0f, 0);
  };

  // ---- SH refinement chain ----
  for (int b = 0; b < 2; ++b) {
    const size_t so = (size_t)b * 7040, fo = (size_t)b * 16384;
    attention(shq + so, 55, f0a + fo, 128, shq + so, sh1 + so);
    attention(f1a + fo, 128, sh1 + so, 55, f1a + fo, F1b + fo);
    attention(sh1 + so, 55, F1b + fo, 128, sh1 + so, sh2 + so);
    attention(f2a + fo, 128, sh2 + so, 55, f2a + fo, F2b + fo);
    attention(sh2 + so, 55, F2b + fo, 128, sh2 + so, sh3 + so);
  }

  // ---- 3x3 SAME convs on features2 (WMMA implicit GEMM) ----
  k_conv3x3_wmma<<<dim3(512, 2), 256, 0, stream>>>(feat2_bf, cdw_bf, pfd, 128, 256);
  k_conv3x3_wmma<<<dim3(512, 2), 256, 0, stream>>>(feat2_bf, cnw_bf, pfn, 128, 256);

  // ---- cosine-sim argmax + gather-multiply (uses original shq) ----
  k_row_l2norm<<<110, 128, 0, stream>>>(shq, shqn, 110);
  for (int b = 0; b < 2; ++b) {
    k_simgather<<<128, 256, 0, stream>>>(shq + (size_t)b * 7040, shqn + (size_t)b * 55,
                                         pfd + (size_t)b * 128 * HW, HW);
    k_simgather<<<128, 256, 0, stream>>>(shq + (size_t)b * 7040, shqn + (size_t)b * 55,
                                         pfn + (size_t)b * 128 * HW, HW);
  }

  // ---- per-pixel transformer heads ----
  auto run_head = [&](const float* pf,
                      const float* m_lng, const float* m_lnb,
                      const unsigned short* w1bf, const float* b1,
                      const unsigned short* w2bf, const float* b2,
                      const float* p1g, const float* p1b,
                      const unsigned short* pw1bf, const float* pb1,
                      const float* p2g, const float* p2b,
                      const unsigned short* pw2bf, const float* pb2,
                      bool normalize, float* outbase) {
    for (int b = 0; b < 2; ++b) {
      k_transpose<<<dim3(1024, 4), dim3(32, 8), 0, stream>>>(pf + (size_t)b * 128 * HW, fbuf, 128, HW);
      attention(fbuf, HW, sh3 + (size_t)b * 7040, 55, fbuf, fbuf);               // f += attn
      lnorm(fbuf, m_lng, m_lnb, sln, HW);
      cast(sln, sln_bf, (size_t)HW * 128);
      gemm(sln_bf, w1bf, b1, 1, nullptr, hbuf, HW, 512, 128, 0, 1.0f, 1);        // gelu
      cast(hbuf, hbuf_bf, (size_t)HW * 512);
      gemm(hbuf_bf, w2bf, b2, 1, fbuf, fbuf, HW, 128, 512, 0, 1.0f, 0);          // f += mlp
      lnorm(fbuf, p1g, p1b, sln, HW);
      cast(sln, sln_bf, (size_t)HW * 128);
      gemm(sln_bf, pw1bf, pb1, 1, nullptr, pbuf, HW, 128, 128, 0, 1.0f, 0);
      lnorm(pbuf, p2g, p2b, sln, HW);
      cast(sln, sln_bf, (size_t)HW * 128);
      if (!normalize) {
        gemm(sln_bf, pw2bf, pb2, 1, nullptr, outbase + (size_t)b * HW, HW, 1, 128, 0, 1.0f, 0);
      } else {
        gemm(sln_bf, pw2bf, pb2, 1, nullptr, obuf, HW, 3, 128, 0, 1.0f, 0);
        k_finalize_normal<<<128, 256, 0, stream>>>(obuf, outbase + (size_t)b * 3 * HW, HW);
      }
    }
  };

  run_head(pfd, md_lng, md_lnb, mdw1_bf, md_b1, mdw2_bf, md_b2,
           pd_ln1g, pd_ln1b, pdw1_bf, pd_b1, pd_ln2g, pd_ln2b, pdw2_bf, pd_b2,
           false, outF);                       // depth -> d_out[0 .. 65536)
  run_head(pfn, mn_lng, mn_lnb, mnw1_bf, mn_b1, mnw2_bf, mn_b2,
           pn_ln1g, pn_ln1b, pnw1_bf, pn_b1, pn_ln2g, pn_ln2b, pnw2_bf, pn_b2,
           true, outF + 65536);                // normal -> d_out[65536 .. 262144)
}
